// GraphAttention_90220083019817
// MI455X (gfx1250) — matrix operand order repaired and checked by reading, not compiled, not run on hardware
//
#include <hip/hip_runtime.h>
#include <hip/hip_bf16.h>

typedef __attribute__((ext_vector_type(16))) _Float16 v16h;
typedef __attribute__((ext_vector_type(8)))  float    v8f;

namespace {
constexpr int B_ = 2, T_ = 5, N_ = 30000, E_ = 180000;
constexpr float EPS_ = 1e-8f;
}

// ---------------- helpers ----------------
__device__ __forceinline__ unsigned enc_f(float x) {
  unsigned u = __float_as_uint(x);
  return (u & 0x80000000u) ? ~u : (u | 0x80000000u);   // order-preserving map
}
__device__ __forceinline__ float dec_f(unsigned u) {
  unsigned v = (u & 0x80000000u) ? (u & 0x7fffffffu) : ~u;
  return __uint_as_float(v);
}

// ---------------- weight packing: f32 (K,Nin) -> f16 transposed padded (Np,Kp) ----------------
__global__ void k_pack_wt(const float* __restrict__ W, int K, int Nin,
                          _Float16* __restrict__ Wt, int Kp, int Np) {
  int idx = blockIdx.x * blockDim.x + threadIdx.x;
  if (idx >= Np * Kp) return;
  int n = idx / Kp, k = idx % Kp;
  float v = (k < K && n < Nin) ? W[(size_t)k * Nin + n] : 0.f;
  Wt[idx] = (_Float16)v;
}

// P[which][l][k][h] = sum_d Wsrc[l][k][32h+d] * A[l][h][d]   (which: 0=src(W),1=dst(W),2=edge(We))
__global__ void k_proj(const float* __restrict__ W, const float* __restrict__ We,
                       const float* __restrict__ Asrc, const float* __restrict__ Adst,
                       const float* __restrict__ Aedge, float* __restrict__ P) {
  int idx = blockIdx.x * blockDim.x + threadIdx.x;
  if (idx >= 3 * 8 * 128 * 4) return;
  int h = idx & 3, k = (idx >> 2) & 127, l = (idx >> 9) & 7, which = idx >> 12;
  const float* Wsrc = (which == 2) ? We : W;
  const float* Av   = (which == 0) ? Asrc : (which == 1) ? Adst : Aedge;
  const float* wr = Wsrc + ((size_t)l * 128 + k) * 128 + h * 32;
  const float* av = Av + ((size_t)l * 4 + h) * 32;
  float s = 0.f;
#pragma unroll
  for (int d = 0; d < 32; ++d) s += wr[d] * av[d];
  P[idx] = s;
}

__global__ void k_pad_bias(const float* __restrict__ b, int n, float* __restrict__ out, int np) {
  int i = threadIdx.x;
  if (i < np) out[i] = (i < n) ? b[i] : 0.f;
}

// ---------------- WMMA GEMM: C[M,Nout] = act(A[M,Kp] @ Wt^T + bias) ----------------
// A: f32 row-major (lda floats/row); Wt: f16, row n holds K-contiguous weights (Kp halves/row).
__global__ __launch_bounds__(256) void k_wmma_gemm(
    const float* __restrict__ A, int lda,
    const _Float16* __restrict__ Wt, int Kp,
    const float* __restrict__ bias,
    float* __restrict__ C, int ldc,
    int M, int colTiles, int tiles, int relu) {
  int tid = blockIdx.x * 8 + (threadIdx.x >> 5);
  if (tid >= tiles) return;                 // wave-uniform: EXEC stays all-1 for WMMA
  int rowTile = tid / colTiles, colTile = tid % colTiles;
  int lane = threadIdx.x & 31;
  int hi = lane >> 4;
  int r  = lane & 15;
  int rowA = rowTile * 16 + r;
  if (rowA >= M) rowA = M - 1;
  const float* ap = A + (size_t)rowA * lda + 16 * hi;
  int n = colTile * 16 + r;
  const _Float16* wp = Wt + (size_t)n * Kp + 16 * hi;
  v8f acc = {0.f, 0.f, 0.f, 0.f, 0.f, 0.f, 0.f, 0.f};
  for (int kk = 0; kk < Kp; kk += 32) {
    // A fragment: lane(0-15)=row r, K = 16*hi + j, contiguous
    const float4 x0 = *(const float4*)(ap + kk);
    const float4 x1 = *(const float4*)(ap + kk + 4);
    const float4 y0 = *(const float4*)(ap + kk + 8);
    const float4 y1 = *(const float4*)(ap + kk + 12);
    v16h a;
    a[0]  = (_Float16)x0.x; a[1]  = (_Float16)x0.y; a[2]  = (_Float16)x0.z; a[3]  = (_Float16)x0.w;
    a[4]  = (_Float16)x1.x; a[5]  = (_Float16)x1.y; a[6]  = (_Float16)x1.z; a[7]  = (_Float16)x1.w;
    a[8]  = (_Float16)y0.x; a[9]  = (_Float16)y0.y; a[10] = (_Float16)y0.z; a[11] = (_Float16)y0.w;
    a[12] = (_Float16)y1.x; a[13] = (_Float16)y1.y; a[14] = (_Float16)y1.z; a[15] = (_Float16)y1.w;
    // B fragment: col n = lane&15, K = kk + 16*hi + j, contiguous halves
    v16h b = *(const v16h*)(wp + kk);
    acc = __builtin_amdgcn_wmma_f32_16x16x32_f16(false, a, false, b, (short)0, acc, false, false);
  }
  float bc = bias ? bias[n] : 0.f;
#pragma unroll
  for (int rr = 0; rr < 8; ++rr) {
    int rowOut = rowTile * 16 + rr + 8 * hi;   // C: VGPR rr -> M = rr + 8*hi, N = lane&15
    if (rowOut < M) {
      float v = acc[rr] + bc;
      if (relu) v = fmaxf(v, 0.f);
      C[(size_t)rowOut * ldc + n] = v;
    }
  }
}

// ---------------- LayerNorm over 128, one wave per row ----------------
__global__ void k_ln128(float* __restrict__ X, const float* __restrict__ g,
                        const float* __restrict__ bb, int M) {
  int row = blockIdx.x * 8 + (threadIdx.x >> 5);
  int lane = threadIdx.x & 31;
  if (row >= M) return;
  float* x = X + (size_t)row * 128;
  float v0 = x[lane], v1 = x[lane + 32], v2 = x[lane + 64], v3 = x[lane + 96];
  float s = v0 + v1 + v2 + v3;
  for (int o = 16; o >= 1; o >>= 1) s += __shfl_xor(s, o, 32);
  float mu = s * (1.f / 128.f);
  float d0 = v0 - mu, d1 = v1 - mu, d2 = v2 - mu, d3 = v3 - mu;
  float q = d0 * d0 + d1 * d1 + d2 * d2 + d3 * d3;
  for (int o = 16; o >= 1; o >>= 1) q += __shfl_xor(q, o, 32);
  float rstd = rsqrtf(q * (1.f / 128.f) + 1e-5f);
  x[lane]      = d0 * rstd * g[lane]      + bb[lane];
  x[lane + 32] = d1 * rstd * g[lane + 32] + bb[lane + 32];
  x[lane + 64] = d2 * rstd * g[lane + 64] + bb[lane + 64];
  x[lane + 96] = d3 * rstd * g[lane + 96] + bb[lane + 96];
}

// ---------------- features ----------------
__global__ void k_node_feat(const float* __restrict__ cur, const float* __restrict__ nt,
                            const float* __restrict__ mean, const float* __restrict__ stdv,
                            float* __restrict__ Vf) {
  int idx = blockIdx.x * blockDim.x + threadIdx.x;
  if (idx >= N_ * 32) return;
  int n = idx >> 5, j = idx & 31;
  float v = 0.f;
  if (j < 3)       v = (cur[n * 3 + j] - mean[j]) / (stdv[j] + EPS_);
  else if (j < 12) v = (nt[(size_t)n * 9 + (j - 3)] - mean[j]) / (stdv[j] + EPS_);
  Vf[idx] = v;
}

__global__ void k_edge_feat(const int* __restrict__ ed, const float* __restrict__ mp,
                            const float* __restrict__ mean, const float* __restrict__ stdv,
                            float* __restrict__ Ef) {
  int idx = blockIdx.x * blockDim.x + threadIdx.x;
  if (idx >= E_ * 32) return;
  int e = idx >> 5, j = idx & 31;
  float v = 0.f;
  if (j < 3) {
    int snd = ed[e * 2], rcv = ed[e * 2 + 1];
    float dx = mp[snd * 2]     - mp[rcv * 2];
    float dy = mp[snd * 2 + 1] - mp[rcv * 2 + 1];
    float t = (j == 0) ? dx : (j == 1) ? dy : sqrtf(dx * dx + dy * dy);
    v = (t - mean[j]) / (stdv[j] + EPS_);
  }
  Ef[idx] = v;
}

// ---------------- GAT attention phase ----------------
__global__ void k_s_node(const float* __restrict__ V, const float* __restrict__ Ps,
                         const float* __restrict__ Pd, float* __restrict__ ssrc,
                         float* __restrict__ sdst) {
  int idx = blockIdx.x * blockDim.x + threadIdx.x;
  if (idx >= N_ * 8) return;
  int q = idx & 7, n = idx >> 3, h = q & 3;
  const float* P = (q < 4) ? Ps : Pd;
  const float* v = V + (size_t)n * 128;
  float s = 0.f;
#pragma unroll 8
  for (int k = 0; k < 128; ++k) s += v[k] * P[k * 4 + h];
  ((q < 4) ? ssrc : sdst)[n * 4 + h] = s;
}

__global__ void k_s_edge(const float* __restrict__ Eh, const float* __restrict__ Pe,
                         float* __restrict__ sedge) {
  int idx = blockIdx.x * blockDim.x + threadIdx.x;
  if (idx >= E_ * 4) return;
  int h = idx & 3, e = idx >> 2;
  const float* v = Eh + (size_t)e * 128;
  float s = 0.f;
#pragma unroll 8
  for (int k = 0; k < 128; ++k) s += v[k] * Pe[k * 4 + h];
  sedge[idx] = s;
}

__global__ void k_edge_logit(const int* __restrict__ ed, const float* __restrict__ ssrc,
                             const float* __restrict__ sdst, const float* __restrict__ sedge,
                             float* __restrict__ logit, unsigned* __restrict__ menc) {
  int idx = blockIdx.x * blockDim.x + threadIdx.x;
  if (idx >= E_ * 4) return;
  int e = idx >> 2, h = idx & 3;
  int snd = ed[e * 2], rcv = ed[e * 2 + 1];
  float lg = ssrc[snd * 4 + h] + sdst[rcv * 4 + h] + sedge[idx];
  lg = (lg >= 0.f) ? lg : 0.2f * lg;           // leaky_relu(0.2)
  logit[idx] = lg;
  atomicMax(&menc[rcv * 4 + h], enc_f(lg));    // segment_max via ordered-uint encoding
}

__global__ void k_edge_exp(const int* __restrict__ ed, float* __restrict__ logit,
                           const unsigned* __restrict__ menc, float* __restrict__ den) {
  int idx = blockIdx.x * blockDim.x + threadIdx.x;
  if (idx >= E_ * 4) return;
  int e = idx >> 2, h = idx & 3;
  int rcv = ed[e * 2 + 1];
  float ex = expf(logit[idx] - dec_f(menc[rcv * 4 + h]));
  logit[idx] = ex;                              // reuse buffer as ex
  atomicAdd(&den[rcv * 4 + h], ex);
}

__global__ void k_edge_msg(const int* __restrict__ ed, const float* __restrict__ ex,
                           const float* __restrict__ den, const float* __restrict__ hbuf,
                           float* __restrict__ msg) {
  int e = blockIdx.x;
  int c = threadIdx.x;                          // 128 channels
  int snd = ed[e * 2], rcv = ed[e * 2 + 1];
  int h = c >> 5;
  float alpha = ex[e * 4 + h] / (den[rcv * 4 + h] + 1e-16f);
  atomicAdd(&msg[(size_t)rcv * 128 + c], alpha * hbuf[(size_t)snd * 128 + c]);
}

__global__ void k_vadd(float* __restrict__ V, const float* __restrict__ msg, int n) {
  int i = blockIdx.x * blockDim.x + threadIdx.x;
  if (i < n) V[i] += msg[i];
}

__global__ void k_zero(float* __restrict__ p, int n) {
  int i = blockIdx.x * blockDim.x + threadIdx.x;
  if (i < n) p[i] = 0.f;
}

// ---------------- integrate step output ----------------
__global__ void k_step_out(const float* __restrict__ decout, float* __restrict__ cur,
                           const float* __restrict__ ntn, const float* __restrict__ truenext,
                           const float* __restrict__ noMean, const float* __restrict__ noStd,
                           float* __restrict__ sh_t1, float* __restrict__ oh_t) {
  int idx = blockIdx.x * blockDim.x + threadIdx.x;
  if (idx >= N_ * 3) return;
  int n = idx / 3, j = idx - n * 3;
  float out = decout[(size_t)n * 16 + j];
  float nxt = cur[idx] + out * noStd[j] + noMean[j];
  const float* t9 = ntn + (size_t)n * 9;
  if (t9[4] == 1.f || t9[6] == 1.f || t9[2] == 1.f) nxt = truenext[idx];
  oh_t[idx] = out;
  sh_t1[idx] = nxt;
  cur[idx] = nxt;
}

__global__ void k_init_state(const float* __restrict__ state, float* __restrict__ cur,
                             float* __restrict__ sh) {
  int idx = blockIdx.x * blockDim.x + threadIdx.x;
  if (idx >= B_ * N_ * 3) return;
  int b = idx / (N_ * 3), r = idx - b * (N_ * 3);
  float v = state[(size_t)b * T_ * N_ * 3 + r];
  cur[idx] = v;
  sh[(size_t)b * T_ * N_ * 3 + r] = v;
}

__global__ void k_target(const float* __restrict__ state, const float* __restrict__ noMean,
                         const float* __restrict__ noStd, float* __restrict__ tgt) {
  int idx = blockIdx.x * blockDim.x + threadIdx.x;
  if (idx >= B_ * (T_ - 1) * N_ * 3) return;
  int j = idx % 3;
  int n = (idx / 3) % N_;
  int t = (idx / (3 * N_)) % (T_ - 1);
  int b = idx / (3 * N_ * (T_ - 1));
  size_t cu = ((size_t)(b * T_ + t) * N_ + n) * 3 + j;
  size_t nx = ((size_t)(b * T_ + t + 1) * N_ + n) * 3 + j;
  tgt[idx] = (state[nx] - state[cu] - noMean[j]) / (noStd[j] + EPS_);
}

// ---------------- host ----------------
extern "C" void kernel_launch(void* const* d_in, const int* in_sizes, int n_in,
                              void* d_out, int out_size, void* d_ws, size_t ws_size,
                              hipStream_t stream) {
  (void)in_sizes; (void)n_in; (void)out_size; (void)ws_size;
  const float* mesh_pos  = (const float*)d_in[0];
  const int*   edges     = (const int*)d_in[1];
  const float* state     = (const float*)d_in[2];
  const float* node_type = (const float*)d_in[3];
  // params: jax pytree sorted-key flatten order
  const float* decW1 = (const float*)d_in[4];
  const float* decW2 = (const float*)d_in[5];
  const float* decW3 = (const float*)d_in[6];
  const float* decb1 = (const float*)d_in[7];
  const float* decb2 = (const float*)d_in[8];
  const float* decb3 = (const float*)d_in[9];
  const float* feW1  = (const float*)d_in[10];
  const float* feW2  = (const float*)d_in[11];
  const float* feW3  = (const float*)d_in[12];
  const float* feb1  = (const float*)d_in[13];
  const float* feb2  = (const float*)d_in[14];
  const float* feb3  = (const float*)d_in[15];
  const float* feLnB = (const float*)d_in[16];
  const float* feLnG = (const float*)d_in[17];
  const float* fvW1  = (const float*)d_in[18];
  const float* fvW2  = (const float*)d_in[19];
  const float* fvW3  = (const float*)d_in[20];
  const float* fvb1  = (const float*)d_in[21];
  const float* fvb2  = (const float*)d_in[22];
  const float* fvb3  = (const float*)d_in[23];
  const float* fvLnB = (const float*)d_in[24];
  const float* fvLnG = (const float*)d_in[25];
  const float* gatW  = (const float*)d_in[26];
  const float* gatWe = (const float*)d_in[27];
  const float* gatAd = (const float*)d_in[28];
  const float* gatAe = (const float*)d_in[29];
  const float* gatAs = (const float*)d_in[30];
  const float* neMean = (const float*)d_in[31];
  const float* neStd  = (const float*)d_in[32];
  const float* nnMean = (const float*)d_in[33];
  const float* nnStd  = (const float*)d_in[34];
  const float* noMean = (const float*)d_in[35];
  const float* noStd  = (const float*)d_in[36];

  float* out_sh = (float*)d_out;                                    // (B,T,N,3)
  float* out_oh = out_sh + (size_t)B_ * T_ * N_ * 3;                // (B,T-1,N,3)
  float* out_tg = out_oh + (size_t)B_ * (T_ - 1) * N_ * 3;          // (B,T-1,N,3)

  // ---- workspace carve (256B aligned) ----
  char* w = (char*)d_ws;
  auto alloc = [&](size_t bytes) { void* p = w; w += (bytes + 255) & ~(size_t)255; return p; };
  _Float16* fvW1t = (_Float16*)alloc(128 * 32 * 2);
  _Float16* fvW2t = (_Float16*)alloc(128 * 128 * 2);
  _Float16* fvW3t = (_Float16*)alloc(128 * 128 * 2);
  _Float16* feW1t = (_Float16*)alloc(128 * 32 * 2);
  _Float16* feW2t = (_Float16*)alloc(128 * 128 * 2);
  _Float16* feW3t = (_Float16*)alloc(128 * 128 * 2);
  _Float16* decW1t = (_Float16*)alloc(128 * 128 * 2);
  _Float16* decW2t = (_Float16*)alloc(128 * 128 * 2);
  _Float16* decW3t = (_Float16*)alloc(16 * 128 * 2);
  _Float16* gatWt  = (_Float16*)alloc((size_t)8 * 128 * 128 * 2);
  float* Pall   = (float*)alloc((size_t)3 * 8 * 128 * 4 * 4);
  float* b3pad  = (float*)alloc(16 * 4);
  float* cur    = (float*)alloc((size_t)B_ * N_ * 3 * 4);
  float* Vf     = (float*)alloc((size_t)N_ * 32 * 4);
  float* Ef     = (float*)alloc((size_t)E_ * 32 * 4);
  float* Vbuf   = (float*)alloc((size_t)N_ * 128 * 4);
  float* hbuf   = (float*)alloc((size_t)N_ * 128 * 4);
  float* msg    = (float*)alloc((size_t)N_ * 128 * 4);   // msg, menc, den contiguous (sizes 256B-mult)
  unsigned* menc = (unsigned*)alloc((size_t)N_ * 4 * 4);
  float* den    = (float*)alloc((size_t)N_ * 4 * 4);
  float* ssrc   = (float*)alloc((size_t)N_ * 4 * 4);
  float* sdst   = (float*)alloc((size_t)N_ * 4 * 4);
  float* sedge  = (float*)alloc((size_t)E_ * 4 * 4);
  float* logit  = (float*)alloc((size_t)E_ * 4 * 4);
  float* big1   = (float*)alloc((size_t)E_ * 128 * 4);
  float* big2   = (float*)alloc((size_t)E_ * 128 * 4);
  float* EhBuf  = (float*)alloc((size_t)E_ * 128 * 4);
  float* decout = (float*)alloc((size_t)N_ * 16 * 4);

  auto cdiv = [](int a, int b) { return (a + b - 1) / b; };
  auto pack = [&](const float* W, int K, int Nin, _Float16* Wt, int Kp, int Np) {
    k_pack_wt<<<cdiv(Np * Kp, 256), 256, 0, stream>>>(W, K, Nin, Wt, Kp, Np);
  };
  auto gemm = [&](const float* A, int lda, const _Float16* Wt, int Kp, const float* bias,
                  float* C, int ldc, int M, int Nout, int relu) {
    int colTiles = Nout / 16;
    int tiles = cdiv(M, 16) * colTiles;
    k_wmma_gemm<<<cdiv(tiles, 8), 256, 0, stream>>>(A, lda, Wt, Kp, bias, C, ldc, M, colTiles, tiles, relu);
  };

  // ---- parameter packing (once per launch; deterministic) ----
  pack(fvW1, 12, 128, fvW1t, 32, 128);
  pack(fvW2, 128, 128, fvW2t, 128, 128);
  pack(fvW3, 128, 128, fvW3t, 128, 128);
  pack(feW1, 3, 128, feW1t, 32, 128);
  pack(feW2, 128, 128, feW2t, 128, 128);
  pack(feW3, 128, 128, feW3t, 128, 128);
  pack(decW1, 128, 128, decW1t, 128, 128);
  pack(decW2, 128, 128, decW2t, 128, 128);
  pack(decW3, 128, 3, decW3t, 128, 16);
  for (int l = 0; l < 8; ++l)
    pack(gatW + (size_t)l * 128 * 128, 128, 128, gatWt + (size_t)l * 128 * 128, 128, 128);
  k_proj<<<cdiv(3 * 8 * 128 * 4, 256), 256, 0, stream>>>(gatW, gatWe, gatAs, gatAd, gatAe, Pall);
  k_pad_bias<<<1, 16, 0, stream>>>(decb3, 3, b3pad, 16);

  // ---- scan-independent outputs ----
  k_init_state<<<cdiv(B_ * N_ * 3, 256), 256, 0, stream>>>(state, cur, out_sh);
  k_target<<<cdiv(B_ * (T_ - 1) * N_ * 3, 256), 256, 0, stream>>>(state, noMean, noStd, out_tg);

  // ---- time scan ----
  for (int t = 0; t < T_ - 1; ++t) {
    for (int b = 0; b < B_; ++b) {
      const float* nt_t   = node_type + ((size_t)(b * T_ + t) * N_) * 9;
      const float* nt_t1  = node_type + ((size_t)(b * T_ + t + 1) * N_) * 9;
      const float* mp_t   = mesh_pos + ((size_t)(b * T_ + t) * N_) * 2;
      const int*   ed_t   = edges + ((size_t)(b * T_ + t) * E_) * 2;
      const float* tnext  = state + ((size_t)(b * T_ + t + 1) * N_) * 3;
      float* curb  = cur + (size_t)b * N_ * 3;
      float* sh_t1 = out_sh + ((size_t)(b * T_ + t + 1) * N_) * 3;
      float* oh_t  = out_oh + ((size_t)(b * (T_ - 1) + t) * N_) * 3;

      // node encoder: 12->128->128->128 + LN
      k_node_feat<<<cdiv(N_ * 32, 256), 256, 0, stream>>>(curb, nt_t, nnMean, nnStd, Vf);
      gemm(Vf, 32, fvW1t, 32, fvb1, big1, 128, N_, 128, 1);
      gemm(big1, 128, fvW2t, 128, fvb2, big2, 128, N_, 128, 1);
      gemm(big2, 128, fvW3t, 128, fvb3, Vbuf, 128, N_, 128, 0);
      k_ln128<<<cdiv(N_, 8), 256, 0, stream>>>(Vbuf, fvLnG, fvLnB, N_);

      // edge encoder: 3->128->128->128 + LN
      k_edge_feat<<<cdiv(E_ * 32, 256), 256, 0, stream>>>(ed_t, mp_t, neMean, neStd, Ef);
      gemm(Ef, 32, feW1t, 32, feb1, big1, 128, E_, 128, 1);
      gemm(big1, 128, feW2t, 128, feb2, big2, 128, E_, 128, 1);
      gemm(big2, 128, feW3t, 128, feb3, EhBuf, 128, E_, 128, 0);
      k_ln128<<<cdiv(E_, 8), 256, 0, stream>>>(EhBuf, feLnG, feLnB, E_);

      // 8 GAT layers with residual
      for (int l = 0; l < 8; ++l) {
        k_zero<<<cdiv(N_ * 136, 256), 256, 0, stream>>>(msg, N_ * 136);  // msg+menc+den
        gemm(Vbuf, 128, gatWt + (size_t)l * 128 * 128, 128, nullptr, hbuf, 128, N_, 128, 0);
        k_s_node<<<cdiv(N_ * 8, 256), 256, 0, stream>>>(Vbuf, Pall + (size_t)(0 * 8 + l) * 512,
                                                        Pall + (size_t)(1 * 8 + l) * 512, ssrc, sdst);
        k_s_edge<<<cdiv(E_ * 4, 256), 256, 0, stream>>>(EhBuf, Pall + (size_t)(2 * 8 + l) * 512, sedge);
        k_edge_logit<<<cdiv(E_ * 4, 256), 256, 0, stream>>>(ed_t, ssrc, sdst, sedge, logit, menc);
        k_edge_exp<<<cdiv(E_ * 4, 256), 256, 0, stream>>>(ed_t, logit, menc, den);
        k_edge_msg<<<E_, 128, 0, stream>>>(ed_t, logit, den, hbuf, msg);
        k_vadd<<<cdiv(N_ * 128, 256), 256, 0, stream>>>(Vbuf, msg, N_ * 128);
      }

      // decoder: 128->128->128->3 (padded to 16)
      gemm(Vbuf, 128, decW1t, 128, decb1, big1, 128, N_, 128, 1);
      gemm(big1, 128, decW2t, 128, decb2, big2, 128, N_, 128, 1);
      gemm(big2, 128, decW3t, 128, b3pad, decout, 16, N_, 16, 0);
      k_step_out<<<cdiv(N_ * 3, 256), 256, 0, stream>>>(decout, curb, nt_t1, tnext,
                                                        noMean, noStd, sh_t1, oh_t);
    }
  }
}
